// AttentionPool_64123861729630
// MI455X (gfx1250) — matrix operand order repaired and checked by reading, not compiled, not run on hardware
//
#include <hip/hip_runtime.h>
#include <hip/hip_bf16.h>

typedef __attribute__((ext_vector_type(16))) __bf16 v16bf;
typedef __attribute__((ext_vector_type(8)))  float  v8f;

// ---------------- ws layout (float offsets) ----------------
// scores  : [0, 131072)            B*T = 16*8192
// bias    : [131072, 131328)       256
// stats   : [131328, 131360)       16 * (max,sum)
// part    : [131360, 262432)       32 chunks * 16 b * 256 f
// w1pk    : [262432, +32768 uints) 256*256 bf16 packed pairs (128 KB)

__device__ __forceinline__ unsigned int bpack(float lo, float hi) {
    // two fp32 -> packed bf16 pair, round-to-nearest-even
    unsigned int ul = __float_as_uint(lo);
    unsigned int uh = __float_as_uint(hi);
    ul += 0x7FFFu + ((ul >> 16) & 1u);
    uh += 0x7FFFu + ((uh >> 16) & 1u);
    return (ul >> 16) | (uh & 0xFFFF0000u);
}

// Hardware tanh on gfx1250 (V_TANH_F32, single TRANS32 op); branch-free fallback.
__device__ __forceinline__ float fast_tanh(float x) {
#if __has_builtin(__builtin_amdgcn_tanhf)
    return __builtin_amdgcn_tanhf(x);
#else
    float ax = __builtin_fabsf(x);
    float e  = __expf(ax + ax);              // overflow -> +inf -> t -> 1
    float t  = 1.f - 2.f / (e + 1.f);
    return __builtin_copysignf(t, x);
#endif
}

// Pack W1 (fp32 row-major [k][n]) into the wave32 WMMA B-operand layout,
// tiled as 16 column-tiles x 8 K-blocks; each tile chunk is 256 u32:
// u32 index = lane*8 + j, holding K = 2j + 8*(lane>>4) + (j>=4 ? 8 : 0) (+ kk*32), N = c*16 + (lane&15).
__global__ __launch_bounds__(256) void pack_w1(const float* __restrict__ W1,
                                               unsigned int* __restrict__ w1pk) {
    int id   = blockIdx.x * 256 + threadIdx.x;   // 32768 threads
    int j    = id & 7;
    int lane = (id >> 3) & 31;
    int kk   = (id >> 8) & 7;
    int c    = id >> 11;                          // 0..15
    int hi   = lane >> 4;
    int n    = c * 16 + (lane & 15);
    int k0   = kk * 32 + 2 * j + hi * 8 + (j & 4) * 2;
    float f0 = W1[k0 * 256 + n];
    float f1 = W1[(k0 + 1) * 256 + n];
    w1pk[(c * 8 + kk) * 256 + lane * 8 + j] = bpack(f0, f1);
}

// bias[n] = sum_k vm[k] * W2[k][n]
__global__ __launch_bounds__(256) void bias_k(const float* __restrict__ W2,
                                              const float* __restrict__ vm,
                                              float* __restrict__ bias) {
    int n = threadIdx.x;
    float acc = 0.f;
    for (int k = 0; k < 256; ++k) acc += vm[k] * W2[k * 256 + n];
    bias[n] = acc;
}

// Fused scores = tanh(X @ W1 + bias) . vm   via bf16 WMMA, f32 accumulate.
// One wave handles one 16-token tile; two column tiles per iteration for
// two independent WMMA accumulation chains (XDL ILP). 8 waves / block.
__global__ __launch_bounds__(256) void score_gemm(const float* __restrict__ x,
                                                  const unsigned int* __restrict__ w1pk,
                                                  const float* __restrict__ bias,
                                                  const float* __restrict__ vm,
                                                  float* __restrict__ scores) {
    const int lane = threadIdx.x & 31;
    const int wave = threadIdx.x >> 5;
    const int tile = blockIdx.x * 8 + wave;      // 0..8191
    const int m  = lane & 15;
    const int hi = lane >> 4;

    const float* xrow = x + ((size_t)tile * 16 + m) * 256;

    union AV { v16bf v; unsigned int u[8]; };

    // Load + convert all eight 16x32 A tiles (ISA A layout):
    // e in [0,8): K = kk*32 + hi*8 + e ; e in [8,16): K = kk*32 + 16 + hi*8 + (e-8)
    AV a[8];
#pragma unroll
    for (int kk = 0; kk < 8; ++kk) {
        const float4* p0 = (const float4*)(xrow + kk * 32 + hi * 8);
        const float4* p1 = (const float4*)(xrow + kk * 32 + 16 + hi * 8);
        float4 r0 = p0[0], r1 = p0[1];
        float4 r2 = p1[0], r3 = p1[1];
        a[kk].u[0] = bpack(r0.x, r0.y); a[kk].u[1] = bpack(r0.z, r0.w);
        a[kk].u[2] = bpack(r1.x, r1.y); a[kk].u[3] = bpack(r1.z, r1.w);
        a[kk].u[4] = bpack(r2.x, r2.y); a[kk].u[5] = bpack(r2.z, r2.w);
        a[kk].u[6] = bpack(r3.x, r3.y); a[kk].u[7] = bpack(r3.z, r3.w);
    }

    float s[8] = {0.f, 0.f, 0.f, 0.f, 0.f, 0.f, 0.f, 0.f};

    for (int c = 0; c < 16; c += 2) {
        v8f d0 = {0.f, 0.f, 0.f, 0.f, 0.f, 0.f, 0.f, 0.f};
        v8f d1 = {0.f, 0.f, 0.f, 0.f, 0.f, 0.f, 0.f, 0.f};
        const unsigned int* base0 = w1pk + (c * 8) * 256 + lane * 8;
#pragma unroll
        for (int kk = 0; kk < 8; ++kk) {
            AV b0, b1;
            const uint4* bp0 = (const uint4*)(base0 + kk * 256);
            const uint4* bp1 = (const uint4*)(base0 + kk * 256 + 2048);
            uint4 q0 = bp0[0], q1 = bp0[1];
            uint4 q2 = bp1[0], q3 = bp1[1];
            b0.u[0] = q0.x; b0.u[1] = q0.y; b0.u[2] = q0.z; b0.u[3] = q0.w;
            b0.u[4] = q1.x; b0.u[5] = q1.y; b0.u[6] = q1.z; b0.u[7] = q1.w;
            b1.u[0] = q2.x; b1.u[1] = q2.y; b1.u[2] = q2.z; b1.u[3] = q2.w;
            b1.u[4] = q3.x; b1.u[5] = q3.y; b1.u[6] = q3.z; b1.u[7] = q3.w;
            d0 = __builtin_amdgcn_wmma_f32_16x16x32_bf16(
                     false, a[kk].v, false, b0.v, (short)0, d0, false, false);
            d1 = __builtin_amdgcn_wmma_f32_16x16x32_bf16(
                     false, a[kk].v, false, b1.v, (short)0, d1, false, false);
        }
        // D layout: VGPR r, lanes0-15: (M=r, N=lane); lanes16-31: (M=r+8, N=lane-16)
        const int n0 = c * 16 + (lane & 15);
        const int n1 = n0 + 16;
        const float bn0 = bias[n0], vn0 = vm[n0];
        const float bn1 = bias[n1], vn1 = vm[n1];
#pragma unroll
        for (int r = 0; r < 8; ++r) {
            s[r] += fast_tanh(d0[r] + bn0) * vn0;
            s[r] += fast_tanh(d1[r] + bn1) * vn1;
        }
    }

    // reduce the 16 N-lanes within each half-wave
#pragma unroll
    for (int r = 0; r < 8; ++r) {
        float v = s[r];
        v += __shfl_xor(v, 1, 16);
        v += __shfl_xor(v, 2, 16);
        v += __shfl_xor(v, 4, 16);
        v += __shfl_xor(v, 8, 16);
        s[r] = v;
    }
    if ((lane & 15) == 0) {
        const int t0 = tile * 16 + hi * 8;
#pragma unroll
        for (int r = 0; r < 8; ++r) scores[t0 + r] = s[r];
    }
}

// Per-batch softmax stats: max and sum of exp over T=8192.
__global__ __launch_bounds__(256) void softmax_stats(const float* __restrict__ scores,
                                                     float* __restrict__ stats) {
    __shared__ float red[256];
    const int b = blockIdx.x;
    const float* s = scores + b * 8192;
    const int tid = threadIdx.x;

    float mx = -3.402823466e38f;
    for (int i = tid; i < 8192; i += 256) mx = fmaxf(mx, s[i]);
    red[tid] = mx; __syncthreads();
    for (int off = 128; off > 0; off >>= 1) {
        if (tid < off) red[tid] = fmaxf(red[tid], red[tid + off]);
        __syncthreads();
    }
    mx = red[0]; __syncthreads();

    float sum = 0.f;
    for (int i = tid; i < 8192; i += 256) sum += __expf(s[i] - mx);
    red[tid] = sum; __syncthreads();
    for (int off = 128; off > 0; off >>= 1) {
        if (tid < off) red[tid] += red[tid + off];
        __syncthreads();
    }
    if (tid == 0) { stats[b * 2] = mx; stats[b * 2 + 1] = red[0]; }
}

// Partial out[b][f] over a 256-token chunk. Each thread owns a feature quad
// (float4 -> global_load_b128), 4 token-substrides per block, LDS reduce.
__global__ __launch_bounds__(256) void weighted_sum(const float* __restrict__ x,
                                                    const float* __restrict__ scores,
                                                    const float* __restrict__ stats,
                                                    float* __restrict__ part) {
    __shared__ float  wsh[256];
    __shared__ float4 racc[256];
    const int b = blockIdx.y, chunk = blockIdx.x;
    const int tid = threadIdx.x;
    const int t0 = chunk * 256;

    const float mx  = stats[b * 2];
    const float inv = 1.f / stats[b * 2 + 1];
    wsh[tid] = __expf(scores[b * 8192 + t0 + tid] - mx) * inv;
    __syncthreads();

    const int fq = tid & 63;    // feature quad: features [4*fq, 4*fq+4)
    const int ts = tid >> 6;    // token substride 0..3

    const float* xb = x + ((size_t)b * 8192 + t0) * 256;
    float4 acc = {0.f, 0.f, 0.f, 0.f};
#pragma unroll 4
    for (int t = ts; t < 256; t += 4) {
        const float w = wsh[t];
        float4 v = *(const float4*)(xb + (size_t)t * 256 + fq * 4);
        acc.x += w * v.x; acc.y += w * v.y;
        acc.z += w * v.z; acc.w += w * v.w;
    }
    racc[tid] = acc;
    __syncthreads();

    if (ts == 0) {
        float4 a0 = racc[fq], a1 = racc[fq + 64], a2 = racc[fq + 128], a3 = racc[fq + 192];
        float4 r;
        r.x = a0.x + a1.x + a2.x + a3.x;
        r.y = a0.y + a1.y + a2.y + a3.y;
        r.z = a0.z + a1.z + a2.z + a3.z;
        r.w = a0.w + a1.w + a2.w + a3.w;
        *(float4*)(part + (chunk * 16 + b) * 256 + fq * 4) = r;
    }
}

__global__ __launch_bounds__(256) void final_reduce(const float* __restrict__ part,
                                                    float* __restrict__ out) {
    const int b = blockIdx.x, f = threadIdx.x;
    float acc = 0.f;
#pragma unroll
    for (int c = 0; c < 32; ++c) acc += part[(c * 16 + b) * 256 + f];
    out[b * 256 + f] = acc;
}

extern "C" void kernel_launch(void* const* d_in, const int* in_sizes, int n_in,
                              void* d_out, int out_size, void* d_ws, size_t ws_size,
                              hipStream_t stream) {
    (void)in_sizes; (void)n_in; (void)out_size; (void)ws_size;
    const float* x  = (const float*)d_in[0];
    const float* W1 = (const float*)d_in[1];
    const float* W2 = (const float*)d_in[2];
    const float* vm = (const float*)d_in[3];
    float* out = (float*)d_out;

    float* wsf          = (float*)d_ws;
    float* scores       = wsf;                 // 131072
    float* bias         = wsf + 131072;        // 256
    float* stats        = wsf + 131328;        // 32
    float* part         = wsf + 131360;        // 131072
    unsigned int* w1pk  = (unsigned int*)(wsf + 262432); // 32768 u32

    pack_w1      <<<128, 256, 0, stream>>>(W1, w1pk);
    bias_k       <<<1,   256, 0, stream>>>(W2, vm, bias);
    score_gemm   <<<1024,256, 0, stream>>>(x, w1pk, bias, vm, scores);
    softmax_stats<<<16,  256, 0, stream>>>(scores, stats);
    weighted_sum <<<dim3(32, 16), 256, 0, stream>>>(x, scores, stats, part);
    final_reduce <<<16,  256, 0, stream>>>(part, out);
}
